// SynthesisInput_43336220017232
// MI455X (gfx1250) — matrix operand order repaired and checked by reading, not compiled, not run on hardware
//
#include <hip/hip_runtime.h>

typedef __attribute__((ext_vector_type(16))) _Float16 v16h;
typedef __attribute__((ext_vector_type(8)))  float    v8f;
typedef int v4i_ __attribute__((vector_size(16)));
typedef __attribute__((address_space(1))) v4i_* gbl128_p;
typedef __attribute__((address_space(3))) v4i_* lds128_p;

#define BATCH   32
#define CH      1024
#define WDIM    512
#define HWSZ    4096
#define LDK     40          // LDS row stride in halfs (32 + 8 pad)
#define TWO_PI  6.283185307179586f

#if defined(__has_builtin)
# if __has_builtin(__builtin_amdgcn_global_load_async_to_lds_b128)
#  define HAVE_ASYNC_COPY 1
# endif
# if __has_builtin(__builtin_amdgcn_s_wait_asynccnt)
#  define HAVE_WAIT_ASYNC 1
# endif
#endif

union Frag16 { v16h v; uint4 q[2]; };

// ---------------------------------------------------------------------------
// Kernel 1: per-batch affine transform -> per-channel (2pi*fr0, 2pi*fr1,
// 2pi*ph, amp) packed as float4 into workspace.
// ---------------------------------------------------------------------------
__global__ __launch_bounds__(256)
void params_kernel(const float* __restrict__ w,
                   const float* __restrict__ aw,
                   const float* __restrict__ ab,
                   const float* __restrict__ freqs,
                   const float* __restrict__ phases,
                   float4* __restrict__ params)
{
    const int b   = blockIdx.x;
    const int tid = threadIdx.x;

    __shared__ float4 red[256];
    __shared__ float  bc[4];

    float4 p = make_float4(0.f, 0.f, 0.f, 0.f);
    for (int i = tid; i < WDIM; i += 256) {
        float wv = w[b * WDIM + i];
        p.x += wv * aw[0 * WDIM + i];
        p.y += wv * aw[1 * WDIM + i];
        p.z += wv * aw[2 * WDIM + i];
        p.w += wv * aw[3 * WDIM + i];
    }
    red[tid] = p;
    __syncthreads();
    for (int s = 128; s > 0; s >>= 1) {
        if (tid < s) {
            float4 o = red[tid + s];
            red[tid].x += o.x; red[tid].y += o.y;
            red[tid].z += o.z; red[tid].w += o.w;
        }
        __syncthreads();
    }
    if (tid == 0) {
        const float gain = 0.044194173824159216f;   // 1/sqrt(512)
        float t0 = red[0].x * gain + ab[0];
        float t1 = red[0].y * gain + ab[1];
        float t2 = red[0].z * gain + ab[2];
        float t3 = red[0].w * gain + ab[3];
        float inv = rsqrtf(t0 * t0 + t1 * t1);
        float cn = t0 * inv, sn = t1 * inv;
        float tx = t2 * inv, ty = t3 * inv;
        bc[0] = cn;
        bc[1] = sn;
        bc[2] = -cn * tx + sn * ty;    // trans0
        bc[3] = -sn * tx - cn * ty;    // trans1
    }
    __syncthreads();
    const float cn = bc[0], sn = bc[1], tr0 = bc[2], tr1 = bc[3];

    for (int c = tid; c < CH; c += 256) {
        float f0 = freqs[c * 2 + 0];
        float f1 = freqs[c * 2 + 1];
        float fr0 =  f0 * cn + f1 * sn;
        float fr1 = -f0 * sn + f1 * cn;
        float ph  = phases[c] + f0 * tr0 + f1 * tr1;
        float r   = sqrtf(fr0 * fr0 + fr1 * fr1);
        float amp = 1.0f - (r - 2.0f) * (1.0f / 30.0f);  // (SR/2 - BW) = 30
        amp = fminf(fmaxf(amp, 0.0f), 1.0f);
        params[b * CH + c] = make_float4(fr0 * TWO_PI, fr1 * TWO_PI,
                                         ph * TWO_PI, amp);
    }
}

// ---------------------------------------------------------------------------
// Kernel 2: convert weight/sqrt(CH) to fp16 once (resident in L2 afterwards).
// ---------------------------------------------------------------------------
__global__ __launch_bounds__(256)
void wconv_kernel(const float* __restrict__ src, _Float16* __restrict__ dst)
{
    int i = blockIdx.x * 256 + threadIdx.x;
    dst[i] = (_Float16)(src[i] * 0.03125f);   // 1/sqrt(1024) = 1/32
}

// ---------------------------------------------------------------------------
// Kernel 3: fused sin-feature synthesis + WMMA GEMM.
//   Block tile: O=128 x M(spatial)=256, K-step 32.
//   8 waves arranged 2(O) x 4(M); each wave owns a 64x64 tile
//   -> 16 WMMAs per 8 fragment loads per K-step.
//   Weight tile staged with GLOBAL_LOAD_ASYNC_TO_LDS (overlapped with the
//   v_sin feature synthesis), feature tile computed in-place into LDS.
// ---------------------------------------------------------------------------
__global__ __launch_bounds__(256)
void synth_gemm(const float4* __restrict__ params,
                const _Float16* __restrict__ wh,
                float* __restrict__ out)
{
    __shared__ __align__(16) _Float16 sW[128 * LDK];   // weight  [o][k]
    __shared__ __align__(16) _Float16 sF[256 * LDK];   // feature [m][k]

    const int tid   = threadIdx.x;
    const int mtile = blockIdx.x;      // 0..15 : 256 spatial positions
    const int otile = blockIdx.y;      // 0..7  : 128 output channels
    const int b     = blockIdx.z;      // 0..31

    // --- feature staging mapping: thread -> fixed channel column
    const int fk  = tid & 31;          // k within tile (= channel - k0)
    const int fm0 = tid >> 5;          // base spatial row, step 8

    // --- weight staging mapping (128 rows x 32 halfs, 16 halfs per thread)
    const int wo = tid >> 1;           // 0..127 output-channel row
    const int wk = (tid & 1) * 16;     // half-row selector
    const _Float16* wsrc = wh + (size_t)(otile * 128 + wo) * CH + wk;
    _Float16* wdst = &sW[wo * LDK + wk];

    // --- wave / lane decomposition (8 waves: 2 in O x 4 in M)
    const int lane = tid & 31;
    const int wv   = tid >> 5;
    const int qo   = wv >> 2;          // 0..1
    const int qm   = wv & 3;           // 0..3
    const int l16  = lane & 15;
    const int lh   = lane >> 4;        // half-wave selector

    v8f acc[4][4] = {};

    for (int k0 = 0; k0 < CH; k0 += 32) {
        __syncthreads();   // protect previous iteration's fragment reads

        // stage fp16 weight tile (128 x 32) -> LDS
#if HAVE_ASYNC_COPY
        __builtin_amdgcn_global_load_async_to_lds_b128(
            (gbl128_p)(wsrc + k0), (lds128_p)wdst, 0, 0);
        __builtin_amdgcn_global_load_async_to_lds_b128(
            (gbl128_p)(wsrc + k0), (lds128_p)wdst, 16, 0);
#else
        {
            const uint4* s = (const uint4*)(wsrc + k0);
            uint4* d = (uint4*)wdst;
            d[0] = s[0];
            d[1] = s[1];
        }
#endif

        // stage sin features (256 x 32) -> LDS; one channel per thread
        // (overlaps with the in-flight async weight copy)
        {
            float4 prm = params[b * CH + k0 + fk];
            #pragma unroll
            for (int i = 0; i < 32; ++i) {
                int m  = fm0 + 8 * i;
                int mg = mtile * 256 + m;
                int hh = mg >> 6;
                int ww = mg & 63;
                float gx = ((float)(2 * ww + 1) * (1.0f / 64.0f) - 1.0f) * 0.5f;
                float gy = ((float)(2 * hh + 1) * (1.0f / 64.0f) - 1.0f) * 0.5f;
                float val = __sinf(gx * prm.x + gy * prm.y + prm.z) * prm.w;
                sF[m * LDK + fk] = (_Float16)val;
            }
        }

#if HAVE_ASYNC_COPY
# if HAVE_WAIT_ASYNC
        __builtin_amdgcn_s_wait_asynccnt(0);
# else
        asm volatile("s_wait_asynccnt 0x0" ::: "memory");
# endif
#endif
        __syncthreads();

        // load fragments per documented wave32 WMMA layouts
        Frag16 af[4], bf[4];
        #pragma unroll
        for (int of = 0; of < 4; ++of) {
            // A (16x32 f16): lane row = l16; K chunks {lh*8..+7} and {+16}
            const _Float16* pa = &sW[(qo * 64 + of * 16 + l16) * LDK + lh * 8];
            af[of].q[0] = *(const uint4*)pa;
            af[of].q[1] = *(const uint4*)(pa + 16);
        }
        #pragma unroll
        for (int mf = 0; mf < 4; ++mf) {
            // B (32x16 f16): lane col = l16; K chunks {lh*8..+7} and {+16}
            const _Float16* pb = &sF[(qm * 64 + mf * 16 + l16) * LDK + lh * 8];
            bf[mf].q[0] = *(const uint4*)pb;
            bf[mf].q[1] = *(const uint4*)(pb + 16);
        }

        #pragma unroll
        for (int of = 0; of < 4; ++of)
            #pragma unroll
            for (int mf = 0; mf < 4; ++mf)
                acc[of][mf] = __builtin_amdgcn_wmma_f32_16x16x32_f16(
                    false, af[of].v, false, bf[mf].v,
                    (short)0, acc[of][mf], false, false);
    }

    // epilogue: D layout -> VGPR r holds row M=r (lanes 0-15) / M=r+8
    // (lanes 16-31), col N = lane%16. Stores land in 64B-contiguous runs.
    #pragma unroll
    for (int of = 0; of < 4; ++of) {
        #pragma unroll
        for (int mf = 0; mf < 4; ++mf) {
            int og = otile * 128 + qo * 64 + of * 16 + lh * 8;
            int mg = mtile * 256 + qm * 64 + mf * 16 + l16;
            float* po = out + ((size_t)b * CH + og) * HWSZ + mg;
            #pragma unroll
            for (int r = 0; r < 8; ++r)
                po[(size_t)r * HWSZ] = acc[of][mf][r];
        }
    }
}

// ---------------------------------------------------------------------------
extern "C" void kernel_launch(void* const* d_in, const int* in_sizes, int n_in,
                              void* d_out, int out_size, void* d_ws, size_t ws_size,
                              hipStream_t stream)
{
    (void)in_sizes; (void)n_in; (void)out_size; (void)ws_size;

    const float* w      = (const float*)d_in[0];
    const float* aw     = (const float*)d_in[1];
    const float* ab     = (const float*)d_in[2];
    const float* freqs  = (const float*)d_in[3];
    const float* phases = (const float*)d_in[4];
    const float* weight = (const float*)d_in[5];

    float4*   params = (float4*)d_ws;                                  // 512 KB
    _Float16* wh     = (_Float16*)((char*)d_ws +
                        (size_t)BATCH * CH * sizeof(float4));          // 2 MB

    params_kernel<<<BATCH, 256, 0, stream>>>(w, aw, ab, freqs, phases, params);
    wconv_kernel<<<(CH * CH) / 256, 256, 0, stream>>>(weight, wh);

    dim3 grid(16, 8, BATCH);   // m-tiles, o-tiles, batch
    synth_gemm<<<grid, 256, 0, stream>>>(params, wh, (float*)d_out);
}
